// OctreeRenderCUDA_15101105012922
// MI455X (gfx1250) — matrix operand order repaired and checked by reading, not compiled, not run on hardware
//
#include <hip/hip_runtime.h>
#include <hip/hip_bf16.h>

typedef _Float16 v16h __attribute__((ext_vector_type(16)));
typedef _Float16 v8h  __attribute__((ext_vector_type(8)));
typedef float    v8f  __attribute__((ext_vector_type(8)));

#define NS 256          // samples per ray (== blockDim.x)
#define KF 72           // feature dim (3 planes x 24 app channels)
#define KP 96           // K padded to multiple of 32 for WMMA
#define PW 256          // plane W
#define PH 256          // plane H
#define PHW (PW*PH)

__device__ __forceinline__ float sigmoidf(float x) { return 1.0f / (1.0f + expf(-x)); }
__device__ __forceinline__ float softplusf(float x) {
    // stable: max(x,0) + log1p(exp(-|x|))  (matches jax.nn.softplus)
    return fmaxf(x, 0.0f) + log1pf(expf(-fabsf(x)));
}

__global__ __launch_bounds__(256) void tensorf_render_kernel(
    const float* __restrict__ rays,          // [N,6]
    const float* __restrict__ density_plane, // [3,16,256,256]
    const float* __restrict__ density_line,  // [3,16,256]
    const float* __restrict__ app_plane,     // [3,24,256,256]
    const float* __restrict__ app_line,      // [3,24,256]
    const float* __restrict__ basis_mat,     // [72,3]
    const float* __restrict__ aabb,          // [2,3]
    float* __restrict__ out,                 // rgb [N,3] then depth [N,1]
    int n_rays)
{
    __shared__ __align__(16) _Float16 s_feat[NS * KP];   // app features, fp16, 48 KB
    __shared__ __align__(16) _Float16 s_basisT[3 * KP];  // basis^T padded, [n][k]
    __shared__ float s_rgb[NS * 3];
    __shared__ float s_wprod[8];
    __shared__ float s_red[8 * 5];

    const int ray  = blockIdx.x;
    const int s    = threadIdx.x;        // sample index along the ray
    const int wave = s >> 5;
    const int lane = s & 31;

    // ---- ray / AABB setup (redundant per thread; cheap, all L2-resident) ----
    const float ox = rays[(size_t)ray * 6 + 0], oy = rays[(size_t)ray * 6 + 1], oz = rays[(size_t)ray * 6 + 2];
    const float dx = rays[(size_t)ray * 6 + 3], dy = rays[(size_t)ray * 6 + 4], dz = rays[(size_t)ray * 6 + 5];
    const float a0x = aabb[0], a0y = aabb[1], a0z = aabb[2];
    const float a1x = aabb[3], a1y = aabb[4], a1z = aabb[5];

    const float vx = (fabsf(dx) < 1e-6f) ? 1e-6f : dx;
    const float vy = (fabsf(dy) < 1e-6f) ? 1e-6f : dy;
    const float vz = (fabsf(dz) < 1e-6f) ? 1e-6f : dz;
    const float rax = (a1x - ox) / vx, ray_ = (a1y - oy) / vy, raz = (a1z - oz) / vz;
    const float rbx = (a0x - ox) / vx, rby  = (a0y - oy) / vy, rbz = (a0z - oz) / vz;
    float tmin = fmaxf(fmaxf(fminf(rax, rbx), fminf(ray_, rby)), fminf(raz, rbz));
    tmin = fmaxf(tmin, 0.05f);
    float tmax = fminf(fminf(fmaxf(rax, rbx), fmaxf(ray_, rby)), fmaxf(raz, rbz));
    tmax = fmaxf(tmax, tmin + 0.001f);

    const float dz_step = (tmax - tmin) / (float)(NS - 1);
    const float z = tmin + dz_step * (float)s;
    const float dist = dz_step;  // diff of linear z is constant (incl. duplicated last)

    const float px = ox + dx * z, py = oy + dy * z, pz = oz + dz * z;
    const float cx = (px - a0x) / (a1x - a0x) * 2.0f - 1.0f;
    const float cy = (py - a0y) / (a1y - a0y) * 2.0f - 1.0f;
    const float cz = (pz - a0z) / (a1z - a0z) * 2.0f - 1.0f;
    const bool inside = (fabsf(cx) <= 1.0f) && (fabsf(cy) <= 1.0f) && (fabsf(cz) <= 1.0f);

    // ---- stage basis_mat^T (padded) into LDS as fp16 ----
    for (int t = s; t < 3 * KP; t += NS) {
        int k = t % KP, nn = t / KP;
        s_basisT[nn * KP + k] = (_Float16)((k < KF) ? basis_mat[k * 3 + nn] : 0.0f);
    }

    // ---- per-sample tri-plane gathers: density scalar + 72 app features ----
    const float c3[3] = {cx, cy, cz};
    const int MU[3] = {0, 0, 1};   // uv[:,0] -> W (x)
    const int MV[3] = {1, 2, 2};   // uv[:,1] -> H (y)
    const int MW[3] = {2, 1, 0};   // line coordinate

    float sigma_feat = 0.0f;
    if (inside) {
        #pragma unroll
        for (int i = 0; i < 3; ++i) {
            const float u = c3[MU[i]], v = c3[MV[i]], wl = c3[MW[i]];
            float x = (u + 1.0f) * 0.5f * (float)(PW - 1);
            int x0 = (int)floorf(x); x0 = min(max(x0, 0), PW - 2);
            const float wx = x - (float)x0;
            float y = (v + 1.0f) * 0.5f * (float)(PH - 1);
            int y0 = (int)floorf(y); y0 = min(max(y0, 0), PH - 2);
            const float wy = y - (float)y0;
            float t1 = (wl + 1.0f) * 0.5f * (float)(PW - 1);
            int t0 = (int)floorf(t1); t0 = min(max(t0, 0), PW - 2);
            const float wt = t1 - (float)t0;

            // density: 16 channels
            {
                const float* P = density_plane + (size_t)i * 16 * PHW + (size_t)y0 * PW + x0;
                const float* L = density_line + (size_t)i * 16 * PW + t0;
                for (int c = 0; c < 16; ++c) {
                    const float* p = P + (size_t)c * PHW;
                    float v00 = p[0], v01 = p[1], v10 = p[PW], v11 = p[PW + 1];
                    float top = v00 + (v01 - v00) * wx;
                    float bot = v10 + (v11 - v10) * wx;
                    float bil = top + (bot - top) * wy;
                    float l0 = L[c * PW], l1 = L[c * PW + 1];
                    sigma_feat += bil * (l0 + (l1 - l0) * wt);
                }
            }
            // appearance: 24 channels -> LDS fp16
            {
                const float* P = app_plane + (size_t)i * 24 * PHW + (size_t)y0 * PW + x0;
                const float* L = app_line + (size_t)i * 24 * PW + t0;
                for (int c = 0; c < 24; ++c) {
                    const float* p = P + (size_t)c * PHW;
                    float v00 = p[0], v01 = p[1], v10 = p[PW], v11 = p[PW + 1];
                    float top = v00 + (v01 - v00) * wx;
                    float bot = v10 + (v11 - v10) * wx;
                    float bil = top + (bot - top) * wy;
                    float l0 = L[c * PW], l1 = L[c * PW + 1];
                    s_feat[s * KP + i * 24 + c] = (_Float16)(bil * (l0 + (l1 - l0) * wt));
                }
            }
        }
    } else {
        for (int k = 0; k < KF; ++k) s_feat[s * KP + k] = (_Float16)0.0f;
    }
    for (int k = KF; k < KP; ++k) s_feat[s * KP + k] = (_Float16)0.0f;  // K padding

    __syncthreads();

    // ---- WMMA: rgb_raw[16x16] = A(16x96 features) x B(96x16 basis, N=0..2 live) ----
    // A layout (16-bit A 16x32): lanes 0-15: M=lane, K-octet 0..7 & 16..23;
    //                            lanes 16-31: M=lane-16, K-octet 8..15 & 24..31.
    // B layout (16-bit B 32x16): lanes 0-15: N=lane, K-octet 0..7 & 16..23; lanes 16-31: K-octet 8..15 & 24..31.
    {
        const int n     = lane & 15;
        const int halfK = (lane >> 4) << 4;  // 0 or 16 (unused)
        const int koff  = (lane >> 4) << 3;  // 0 or 8  (A)

        v16h Bm[3];
        #pragma unroll
        for (int kb3 = 0; kb3 < 3; ++kb3) {
            v16h b = {};
            if (n < 3) {
                const v8h* q = (const v8h*)&s_basisT[n * KP + kb3 * 32 + koff];
                v8h b0 = q[0], b1 = q[2];
                #pragma unroll
                for (int e = 0; e < 8; ++e) { b[e] = b0[e]; b[e + 8] = b1[e]; }
            }
            Bm[kb3] = b;
        }

        #pragma unroll
        for (int g = 0; g < 2; ++g) {
            const int pt = wave * 32 + g * 16 + n;   // sample row M of this tile
            v8f acc = {};
            #pragma unroll
            for (int kb3 = 0; kb3 < 3; ++kb3) {
                const v8h* pa = (const v8h*)&s_feat[pt * KP + kb3 * 32 + koff];
                const v8h* pb = (const v8h*)&s_feat[pt * KP + kb3 * 32 + 16 + koff];
                v8h a0 = pa[0], a1 = pb[0];
                v16h a;
                #pragma unroll
                for (int e = 0; e < 8; ++e) { a[e] = a0[e]; a[e + 8] = a1[e]; }
                acc = __builtin_amdgcn_wmma_f32_16x16x32_f16(
                    false, a, false, Bm[kb3], (short)0, acc, false, false);
            }
            // D layout: VGPR r, lanes 0-15 -> M=r, lanes 16-31 -> M=r+8; N=lane%16
            const int mbase = wave * 32 + g * 16 + ((lane >= 16) ? 8 : 0);
            if (n < 3) {
                #pragma unroll
                for (int r = 0; r < 8; ++r)
                    s_rgb[(mbase + r) * 3 + n] = sigmoidf(acc[r]);
            }
        }
    }

    __syncthreads();

    // ---- per-ray compositing: transmittance scan + weighted reductions ----
    const float sigma = inside ? softplusf(sigma_feat) : 0.0f;
    const float alpha = 1.0f - expf(-sigma * dist);
    const float f = 1.0f - alpha + 1e-10f;

    // intra-wave inclusive product scan (wave32)
    float p = f;
    #pragma unroll
    for (int off = 1; off < 32; off <<= 1) {
        float t = __shfl_up(p, off);
        if (lane >= off) p *= t;
    }
    float excl = __shfl_up(p, 1);
    if (lane == 0) excl = 1.0f;
    if (lane == 31) s_wprod[wave] = p;   // wave-total product
    __syncthreads();

    float wpre = 1.0f;
    for (int w = 0; w < 8; ++w) if (w < wave) wpre *= s_wprod[w];

    const float T = wpre * excl;           // exclusive cumprod -> transmittance
    const float weight = alpha * T;

    const float r0 = s_rgb[s * 3 + 0], g0 = s_rgb[s * 3 + 1], b0 = s_rgb[s * 3 + 2];
    float vals[5] = { weight, weight * z, weight * r0, weight * g0, weight * b0 };
    #pragma unroll
    for (int j = 0; j < 5; ++j) {
        #pragma unroll
        for (int off = 16; off >= 1; off >>= 1)
            vals[j] += __shfl_xor(vals[j], off);
    }
    if (lane == 0) {
        #pragma unroll
        for (int j = 0; j < 5; ++j) s_red[wave * 5 + j] = vals[j];
    }
    __syncthreads();

    if (s == 0) {
        float S[5] = {0, 0, 0, 0, 0};
        for (int w = 0; w < 8; ++w)
            for (int j = 0; j < 5; ++j) S[j] += s_red[w * 5 + j];
        const float acc_w = S[0];
        const float depth = S[1];
        float rr = S[2] + (1.0f - acc_w);   // WHITE_BG
        float gg = S[3] + (1.0f - acc_w);
        float bb = S[4] + (1.0f - acc_w);
        rr = fminf(fmaxf(rr, 0.0f), 1.0f);
        gg = fminf(fmaxf(gg, 0.0f), 1.0f);
        bb = fminf(fmaxf(bb, 0.0f), 1.0f);
        out[(size_t)ray * 3 + 0] = rr;
        out[(size_t)ray * 3 + 1] = gg;
        out[(size_t)ray * 3 + 2] = bb;
        out[(size_t)n_rays * 3 + ray] = depth;
    }
}

extern "C" void kernel_launch(void* const* d_in, const int* in_sizes, int n_in,
                              void* d_out, int out_size, void* d_ws, size_t ws_size,
                              hipStream_t stream) {
    const float* rays          = (const float*)d_in[0];
    const float* density_plane = (const float*)d_in[1];
    const float* density_line  = (const float*)d_in[2];
    const float* app_plane     = (const float*)d_in[3];
    const float* app_line      = (const float*)d_in[4];
    const float* basis_mat     = (const float*)d_in[5];
    const float* aabb          = (const float*)d_in[6];
    // d_in[7] = N_samples (== 256, fixed by the harness setup)

    const int n_rays = in_sizes[0] / 6;
    tensorf_render_kernel<<<dim3(n_rays), dim3(NS), 0, stream>>>(
        rays, density_plane, density_line, app_plane, app_line,
        basis_mat, aabb, (float*)d_out, n_rays);
    (void)d_ws; (void)ws_size; (void)n_in; (void)out_size;
}
